// MobiusGRU_3865470567043
// MI455X (gfx1250) — compile-verified, matrix operand order checked by reading, not run on hardware
//
#include <hip/hip_runtime.h>
#include <stdint.h>

#define T_DIM 256
#define B_DIM 64
#define I_DIM 512
#define H_DIM 512
#define EPSV 1e-5f
#define MINV 1e-15f

typedef __bf16 bf16t;
typedef bf16t v16bf __attribute__((ext_vector_type(16)));
typedef float v8f   __attribute__((ext_vector_type(8)));

// ---------- scalar helpers ----------
__device__ inline float wred(float v) {
#pragma unroll
  for (int o = 16; o > 0; o >>= 1) v += __shfl_xor(v, o, 32);
  return v;
}
__device__ inline unsigned short f2bf(float f) {
  unsigned int u = __float_as_uint(f);
  u += 0x7fffu + ((u >> 16) & 1u);           // round-to-nearest-even
  return (unsigned short)(u >> 16);
}
__device__ inline float artanh_c(float x) {  // clipped artanh
  x = fminf(fmaxf(x, -1.0f + EPSV), 1.0f - EPSV);
  return 0.5f * (log1pf(x) - log1pf(-x));
}
__device__ inline float tanh15(float x) {
  return tanhf(fminf(fmaxf(x, -15.0f), 15.0f));
}

// ---------- row-parallel Mobius ops (one wave owns one 512-dim row; 16 elems/lane) ----------
__device__ inline void mobius_add_row(const float* x, const float* y, float* o) {
  float x2 = 0.f, y2 = 0.f, xy = 0.f;
#pragma unroll
  for (int j = 0; j < 16; j++) { x2 += x[j] * x[j]; y2 += y[j] * y[j]; xy += x[j] * y[j]; }
  x2 = wred(x2); y2 = wred(y2); xy = wred(xy);
  float c1 = 1.f + 2.f * xy + y2;
  float c2 = 1.f - x2;
  float den = fmaxf(1.f + 2.f * xy + x2 * y2, MINV);
  float inv = 1.f / den;
#pragma unroll
  for (int j = 0; j < 16; j++) o[j] = (c1 * x[j] + c2 * y[j]) * inv;
}
__device__ inline void mobius_res_row(const float* mx, float xn, float* o) {
  float s = 0.f;
#pragma unroll
  for (int j = 0; j < 16; j++) s += mx[j] * mx[j];
  s = wred(s);
  float mxn = fmaxf(sqrtf(s), MINV);
  float scale = (s == 0.f) ? 0.f : tanh15(mxn / xn * artanh_c(xn)) / mxn;
#pragma unroll
  for (int j = 0; j < 16; j++) o[j] = scale * mx[j];
}
__device__ inline void logmap0_row(const float* y, float* o) {
  float s = 0.f;
#pragma unroll
  for (int j = 0; j < 16; j++) s += y[j] * y[j];
  s = wred(s);
  float yn = fmaxf(sqrtf(s), MINV);
  float f = artanh_c(yn) / yn;
#pragma unroll
  for (int j = 0; j < 16; j++) o[j] = y[j] * f;
}

// ---------- WMMA fragment load (bf16, 16x32 A / 32x16 B; ISA 7.12.2 layout) ----------
__device__ inline v16bf load_frag(const unsigned short* base, int r0, int kb, int ld) {
  int lane = threadIdx.x & 31;
  const unsigned short* p = base + (size_t)(r0 + (lane & 15)) * ld + kb + ((lane >> 4) << 3);
  union { uint4 u[2]; v16bf v; } f;
  f.u[0] = *reinterpret_cast<const uint4*>(p);       // K = k..k+7
  f.u[1] = *reinterpret_cast<const uint4*>(p + 16);  // K = k+16..k+23
  return f.v;
}

// 16 rows x 512 cols strip GEMM into LDS (block = 256 threads = 8 waves, 4 N-tiles/wave).
// Depth-3 software pipeline with a 3-slot rotating fragment buffer; sched_group_barrier
// (0x20 = VMEM read, 0x08 = MFMA/WMMA) forces the pattern
//   [10 vmem k0][10 vmem k1] { [10 vmem k_{i+2}][4 wmma k_i] } ...
// so ~20 loads stay in flight past each consumed group and waits become partial
// (s_wait_loadcnt ~20) instead of a full drain before every WMMA.
template <int K>
__device__ inline void gemm_strip(const unsigned short* A, int row0, int lda,
                                  const unsigned short* Bm, int ldb, float* Msh) {
  int wave = threadIdx.x >> 5, lane = threadIdx.x & 31;
  int n0 = wave * 64;
  v8f a0 = {0,0,0,0,0,0,0,0}, a1 = a0, a2 = a0, a3 = a0;

  v16bf fa[3], f0[3], f1[3], f2[3], f3[3];
#pragma unroll
  for (int s = 0; s < 2; s++) {
    fa[s] = load_frag(A,  row0,    s * 32, lda);
    f0[s] = load_frag(Bm, n0,      s * 32, ldb);
    f1[s] = load_frag(Bm, n0 + 16, s * 32, ldb);
    f2[s] = load_frag(Bm, n0 + 32, s * 32, ldb);
    f3[s] = load_frag(Bm, n0 + 48, s * 32, ldb);
  }
  __builtin_amdgcn_sched_group_barrier(0x0020, 10, 0);   // k0 fragment loads
  __builtin_amdgcn_sched_group_barrier(0x0020, 10, 0);   // k1 fragment loads

#pragma unroll
  for (int i = 0; i < K / 32; i++) {
    const int s  = i % 3;
    const int sp = (i + 2) % 3;
    const int kp = i * 32 + 64;
    if (kp < K) {                       // compile-time under full unroll
      fa[sp] = load_frag(A,  row0,    kp, lda);
      f0[sp] = load_frag(Bm, n0,      kp, ldb);
      f1[sp] = load_frag(Bm, n0 + 16, kp, ldb);
      f2[sp] = load_frag(Bm, n0 + 32, kp, ldb);
      f3[sp] = load_frag(Bm, n0 + 48, kp, ldb);
    }
    a0 = __builtin_amdgcn_wmma_f32_16x16x32_bf16(false, fa[s], false, f0[s], (short)0, a0, false, false);
    a1 = __builtin_amdgcn_wmma_f32_16x16x32_bf16(false, fa[s], false, f1[s], (short)0, a1, false, false);
    a2 = __builtin_amdgcn_wmma_f32_16x16x32_bf16(false, fa[s], false, f2[s], (short)0, a2, false, false);
    a3 = __builtin_amdgcn_wmma_f32_16x16x32_bf16(false, fa[s], false, f3[s], (short)0, a3, false, false);
    if (kp < K) __builtin_amdgcn_sched_group_barrier(0x0020, 10, 0);  // loads for k_{i+2}
    __builtin_amdgcn_sched_group_barrier(0x0008, 4, 0);               // 4 WMMAs for k_i
  }

  int mb = (lane >> 4) * 8, nc = lane & 15;
#pragma unroll
  for (int v = 0; v < 8; v++) {
    Msh[(mb + v) * 512 + n0      + nc] = a0[v];
    Msh[(mb + v) * 512 + n0 + 16 + nc] = a1[v];
    Msh[(mb + v) * 512 + n0 + 32 + nc] = a2[v];
    Msh[(mb + v) * 512 + n0 + 48 + nc] = a3[v];
  }
}

// ---------- prep kernels ----------
__global__ void __launch_bounds__(256) k_conv_w(const float* wih, const float* whh,
                                                unsigned short* wihbf, unsigned short* whhbf) {
  int i = blockIdx.x * 256 + threadIdx.x;
  int n = 3 * H_DIM * I_DIM;
  if (i < n) { wihbf[i] = f2bf(wih[i]); whhbf[i] = f2bf(whh[i]); }
}

__global__ void __launch_bounds__(256) k_prep_x(const float* x, unsigned short* xbf, float* xn) {
  int row = blockIdx.x, tid = threadIdx.x;
  const float* xr = x + (size_t)row * I_DIM;
  float v0 = xr[tid], v1 = xr[tid + 256];
  xbf[(size_t)row * I_DIM + tid]       = f2bf(v0);
  xbf[(size_t)row * I_DIM + tid + 256] = f2bf(v1);
  float s = wred(v0 * v0 + v1 * v1);
  __shared__ float ps[8];
  if ((tid & 31) == 0) ps[tid >> 5] = s;
  __syncthreads();
  if (tid == 0) {
    float t = 0.f;
    for (int k = 0; k < 8; k++) t += ps[k];
    xn[row] = fmaxf(sqrtf(t), MINV);
  }
}

__global__ void __launch_bounds__(256) k_init_h(const float* h0, float* hxf,
                                                unsigned short* hxbf, float* hxn) {
  int row = blockIdx.x, tid = threadIdx.x;
  const float* hr = h0 + (size_t)row * H_DIM;
  float v0 = hr[tid], v1 = hr[tid + 256];
  hxf[(size_t)row * H_DIM + tid] = v0;       hxf[(size_t)row * H_DIM + tid + 256] = v1;
  hxbf[(size_t)row * H_DIM + tid] = f2bf(v0); hxbf[(size_t)row * H_DIM + tid + 256] = f2bf(v1);
  float s = wred(v0 * v0 + v1 * v1);
  __shared__ float ps[8];
  if ((tid & 31) == 0) ps[tid >> 5] = s;
  __syncthreads();
  if (tid == 0) {
    float t = 0.f;
    for (int k = 0; k < 8; k++) t += ps[k];
    hxn[row] = fmaxf(sqrtf(t), MINV);
  }
}

// ---------- hoisted input-side mobius_matvec for all T (parallel, WMMA) ----------
__global__ void __launch_bounds__(256) k_input_gemm(const unsigned short* xbf, const float* xn,
                                                    const unsigned short* wihbf, float* P) {
  __shared__ float Msh[16 * 512];
  int strip = blockIdx.x;  // 0..1023  (16 rows of the 16384 flattened (t,b) rows)
  int q = blockIdx.y;      // 0:W_ir 1:W_ih 2:W_iz
  gemm_strip<I_DIM>(xbf, strip * 16, I_DIM, wihbf + (size_t)q * H_DIM * I_DIM, I_DIM, Msh);
  __syncthreads();
  int wave = threadIdx.x >> 5, lane = threadIdx.x & 31;
  for (int rr = wave * 2; rr < wave * 2 + 2; rr++) {
    int rg = strip * 16 + rr;
    float m[16];
#pragma unroll
    for (int j = 0; j < 16; j++) m[j] = Msh[rr * 512 + lane + 32 * j];
    mobius_res_row(m, xn[rg], m);
    float* dst = P + ((size_t)q * T_DIM * B_DIM + rg) * H_DIM;
#pragma unroll
    for (int j = 0; j < 16; j++) dst[lane + 32 * j] = m[j];
  }
}

// ---------- recurrent phase A: r and z gates (grid 4 x 2) ----------
__global__ void __launch_bounds__(256) k_phaseA(const unsigned short* hxbf, const float* hxf,
                                                const float* hxn, const unsigned short* whhbf,
                                                const float* P, const float* bias,
                                                float* zbuf, unsigned short* gbf, int t) {
  __shared__ float Msh[16 * 512];
  int strip = blockIdx.x;        // 0..3 (batch rows)
  int sel = blockIdx.y;          // 0 -> r gate, 1 -> z gate
  int q = sel ? 2 : 0;           // W_hr / W_hz plane; bias rows 0 / 2; P planes 0 / 2
  gemm_strip<H_DIM>(hxbf, strip * 16, H_DIM, whhbf + (size_t)q * H_DIM * H_DIM, H_DIM, Msh);
  __syncthreads();
  int wave = threadIdx.x >> 5, lane = threadIdx.x & 31;
  for (int rr = wave * 2; rr < wave * 2 + 2; rr++) {
    int rg = strip * 16 + rr;
    float m[16], p[16];
#pragma unroll
    for (int j = 0; j < 16; j++) m[j] = Msh[rr * 512 + lane + 32 * j];
    mobius_res_row(m, hxn[rg], m);                       // a = mobius_res(hx@W^T, hx)
    const float* pp = P + ((size_t)q * T_DIM * B_DIM + (size_t)t * B_DIM + rg) * H_DIM;
#pragma unroll
    for (int j = 0; j < 16; j++) p[j] = pp[lane + 32 * j];
    mobius_add_row(m, p, m);                             // + input-side matvec
    const float* bb = bias + (size_t)q * H_DIM;
#pragma unroll
    for (int j = 0; j < 16; j++) p[j] = bb[lane + 32 * j];
    mobius_add_row(m, p, m);                             // + bias
    logmap0_row(m, m);                                   // logmap0 -> pre-activation
    if (sel == 0) {
      const float* hr = hxf + (size_t)rg * H_DIM;
      unsigned short* go = gbf + (size_t)rg * H_DIM;
#pragma unroll
      for (int j = 0; j < 16; j++) {
        float gate = 1.f / (1.f + expf(-m[j]));
        go[lane + 32 * j] = f2bf(gate * hr[lane + 32 * j]);   // g = r * hx (bf16 for GEMM)
      }
    } else {
      float* zo = zbuf + (size_t)rg * H_DIM;
#pragma unroll
      for (int j = 0; j < 16; j++) zo[lane + 32 * j] = 1.f / (1.f + expf(-m[j]));
    }
  }
}

// ---------- recurrent phase B: candidate + state update (grid 4) ----------
__global__ void __launch_bounds__(256) k_phaseB(const unsigned short* gbf, float* hxf,
                                                unsigned short* hxbf, float* hxn,
                                                const unsigned short* whhbf, const float* P,
                                                const float* bias, const float* zbuf,
                                                float* outp, int t) {
  __shared__ float Msh[16 * 512];
  int strip = blockIdx.x;
  gemm_strip<H_DIM>(gbf, strip * 16, H_DIM, whhbf + (size_t)H_DIM * H_DIM, H_DIM, Msh);
  __syncthreads();
  int wave = threadIdx.x >> 5, lane = threadIdx.x & 31;
  for (int rr = wave * 2; rr < wave * 2 + 2; rr++) {
    int rg = strip * 16 + rr;
    float m[16], p[16], hx[16];
#pragma unroll
    for (int j = 0; j < 16; j++) m[j] = Msh[rr * 512 + lane + 32 * j];
    mobius_res_row(m, hxn[rg], m);                        // Wh = mobius_res((r*hx)@Whh^T, hx)
    const float* pp = P + ((size_t)1 * T_DIM * B_DIM + (size_t)t * B_DIM + rg) * H_DIM;
#pragma unroll
    for (int j = 0; j < 16; j++) p[j] = pp[lane + 32 * j];
    mobius_add_row(m, p, m);
    const float* bb = bias + H_DIM;                       // b_h
#pragma unroll
    for (int j = 0; j < 16; j++) p[j] = bb[lane + 32 * j];
    mobius_add_row(m, p, m);                              // m = h_tilde
    const float* hr = hxf + (size_t)rg * H_DIM;
#pragma unroll
    for (int j = 0; j < 16; j++) { hx[j] = hr[lane + 32 * j]; p[j] = -hx[j]; }
    mobius_add_row(p, m, m);                              // m = delta = (-hx) (+) h_tilde
    float d2 = 0.f;
#pragma unroll
    for (int j = 0; j < 16; j++) d2 += m[j] * m[j];
    d2 = wred(d2);
    float dn = fmaxf(sqrtf(d2), MINV);
    const float* zr = zbuf + (size_t)rg * H_DIM;
#pragma unroll
    for (int j = 0; j < 16; j++) p[j] = zr[lane + 32 * j] * m[j];   // z * delta
    mobius_res_row(p, dn, p);                             // pm = mobius_pointwise_mul(z, delta)
    mobius_add_row(hx, p, p);                             // h_new = hx (+) pm
    float hn2 = 0.f;
#pragma unroll
    for (int j = 0; j < 16; j++) hn2 += p[j] * p[j];
    hn2 = wred(hn2);
    float* orow = outp + ((size_t)t * B_DIM + rg) * H_DIM;
    float* hrow = hxf + (size_t)rg * H_DIM;
    unsigned short* hb = hxbf + (size_t)rg * H_DIM;
#pragma unroll
    for (int j = 0; j < 16; j++) {
      int c = lane + 32 * j;
      orow[c] = p[j]; hrow[c] = p[j]; hb[c] = f2bf(p[j]);
    }
    if (lane == 0) hxn[rg] = fmaxf(sqrtf(hn2), MINV);
  }
}

__global__ void __launch_bounds__(256) k_final(const float* hxf, float* dst) {
  int i = blockIdx.x * 256 + threadIdx.x;
  dst[i] = hxf[i];
}

extern "C" void kernel_launch(void* const* d_in, const int* in_sizes, int n_in,
                              void* d_out, int out_size, void* d_ws, size_t ws_size,
                              hipStream_t stream) {
  (void)in_sizes; (void)n_in; (void)out_size; (void)ws_size;
  const float* x    = (const float*)d_in[0];   // [T,B,I]
  const float* h0   = (const float*)d_in[1];   // [B,H]
  const float* wih  = (const float*)d_in[2];   // [3H,I]
  const float* whh  = (const float*)d_in[3];   // [3H,H]
  const float* bias = (const float*)d_in[4];   // [3,H]
  float* out = (float*)d_out;                  // [T,B,H] ++ [1,B,H]

  char* w = (char*)d_ws;
  auto carve = [&](size_t bytes) { void* r = (void*)w; w += (bytes + 255) & ~(size_t)255; return r; };
  unsigned short* wihbf = (unsigned short*)carve((size_t)3 * H_DIM * I_DIM * 2);
  unsigned short* whhbf = (unsigned short*)carve((size_t)3 * H_DIM * H_DIM * 2);
  unsigned short* xbf   = (unsigned short*)carve((size_t)T_DIM * B_DIM * I_DIM * 2);
  float*          xn    = (float*)carve((size_t)T_DIM * B_DIM * 4);
  float*          P     = (float*)carve((size_t)3 * T_DIM * B_DIM * H_DIM * 4);
  float*          hxf   = (float*)carve((size_t)B_DIM * H_DIM * 4);
  unsigned short* hxbf  = (unsigned short*)carve((size_t)B_DIM * H_DIM * 2);
  float*          hxn   = (float*)carve((size_t)B_DIM * 4);
  unsigned short* gbf   = (unsigned short*)carve((size_t)B_DIM * H_DIM * 2);
  float*          zbuf  = (float*)carve((size_t)B_DIM * H_DIM * 4);

  k_conv_w<<<(3 * H_DIM * I_DIM) / 256, 256, 0, stream>>>(wih, whh, wihbf, whhbf);
  k_prep_x<<<T_DIM * B_DIM, 256, 0, stream>>>(x, xbf, xn);
  k_init_h<<<B_DIM, 256, 0, stream>>>(h0, hxf, hxbf, hxn);
  k_input_gemm<<<dim3(T_DIM * B_DIM / 16, 3), 256, 0, stream>>>(xbf, xn, wihbf, P);

  for (int t = 0; t < T_DIM; t++) {
    k_phaseA<<<dim3(B_DIM / 16, 2), 256, 0, stream>>>(hxbf, hxf, hxn, whhbf, P, bias, zbuf, gbf, t);
    k_phaseB<<<dim3(B_DIM / 16, 1), 256, 0, stream>>>(gbf, hxf, hxbf, hxn, whhbf, P, bias, zbuf, out, t);
  }
  k_final<<<(B_DIM * H_DIM) / 256, 256, 0, stream>>>(hxf, out + (size_t)T_DIM * B_DIM * H_DIM);
}
